// Decoder_15092515078764
// MI455X (gfx1250) — matrix operand order repaired and checked by reading, not compiled, not run on hardware
//
#include <hip/hip_runtime.h>
#include <hip/hip_bf16.h>
#include <math.h>

#define C     1024
#define L     2048
#define NWG   64
#define TPB   256
#define PITCH 1032            // LDS weight row pitch in elements (1024 + 8 pad -> conflict-free)
#define NJOBS 36              // 9 row-tiles * 4 k-quarters

typedef __attribute__((ext_vector_type(16))) __bf16 v16bf;
typedef __attribute__((ext_vector_type(8)))  __bf16 v8bf;
typedef __attribute__((ext_vector_type(8)))  float  v8f;

union BfPack { v16bf v; v8bf h[2]; };

__device__ __forceinline__ float sigf(float x) { return 1.0f / (1.0f + __expf(-x)); }

// ---------------- setup kernels (run once per launch) ----------------

// logits s[t] = x0[t] . attn_w[C:2C] + attn_b  (the out-dependent term is constant
// over t and cancels in softmax -> attention weights identical for every timestep)
__global__ void k_scores(const float* __restrict__ x, const float* __restrict__ attn_w,
                         const float* __restrict__ attn_b, float* __restrict__ s) {
  int t = blockIdx.x * 8 + (threadIdx.x >> 5);
  int lane = threadIdx.x & 31;
  const float* row = x + (size_t)t * C;
  const float* w = attn_w + C;
  float acc = 0.f;
  for (int i = lane; i < C; i += 32) acc += row[i] * w[i];
  for (int off = 16; off; off >>= 1) acc += __shfl_xor(acc, off, 32);
  if (lane == 0) s[t] = acc + attn_b[0];
}

__global__ void k_softmax(const float* __restrict__ s, float* __restrict__ a) {
  __shared__ float red[256];
  int tid = threadIdx.x;
  float m = -1e30f;
  for (int i = tid; i < L; i += 256) m = fmaxf(m, s[i]);
  red[tid] = m; __syncthreads();
  for (int off = 128; off; off >>= 1) { if (tid < off) red[tid] = fmaxf(red[tid], red[tid + off]); __syncthreads(); }
  m = red[0]; __syncthreads();
  float sum = 0.f;
  for (int i = tid; i < L; i += 256) sum += __expf(s[i] - m);
  red[tid] = sum; __syncthreads();
  for (int off = 128; off; off >>= 1) { if (tid < off) red[tid] += red[tid + off]; __syncthreads(); }
  float inv = 1.0f / red[0];
  for (int i = tid; i < L; i += 256) a[i] = __expf(s[i] - m) * inv;
}

__global__ void k_av(const float* __restrict__ x, const float* __restrict__ a,
                     float* __restrict__ avf) {
  int c = blockIdx.x * 256 + threadIdx.x;
  float acc = 0.f;
  for (int t = 0; t < L; ++t) acc += a[t] * x[(size_t)t * C + c];
  avf[c] = acc;
}

__global__ void k_gi0(const float* __restrict__ w_ih0, const float* __restrict__ b_ih0,
                      const float* __restrict__ avf, float* __restrict__ gi0) {
  int r = blockIdx.x * 8 + (threadIdx.x >> 5);
  int lane = threadIdx.x & 31;
  const float* row = w_ih0 + (size_t)r * C;
  float acc = 0.f;
  for (int i = lane; i < C; i += 32) acc += row[i] * avf[i];
  for (int off = 16; off; off >>= 1) acc += __shfl_xor(acc, off, 32);
  if (lane == 0) gi0[r] = acc + b_ih0[r];
}

__global__ void k_cvt(const float* __restrict__ w, __bf16* __restrict__ dst, int n) {
  int i = (blockIdx.x * 256 + threadIdx.x) * 4;
  if (i < n) {
    dst[i + 0] = (__bf16)w[i + 0];
    dst[i + 1] = (__bf16)w[i + 1];
    dst[i + 2] = (__bf16)w[i + 2];
    dst[i + 3] = (__bf16)w[i + 3];
  }
}

// ---------------- persistent recurrent kernel ----------------
// Phase t (t = -1..L-1): computes h1^{t+1} = gru1(h0^{t+1}, h1^t) and
// h0^{t+2} = gru0(av, h0^{t+1}) -> one device barrier per step.
// WG wg owns output elements [wg*16, wg*16+16) of both layers; its 144 weight
// rows (3 gates x {w_hh0, w_ih1, w_hh1}) live in LDS for the whole kernel.
// Each wave has a fixed k-quarter (wave&3) and tile parity (wave>>2), so its
// A-operand (replicated h) is preloaded ONCE into 64 VGPRs per h-vector and
// reused across all its tiles: inner loop = 2x ds_load_b128 (B) + 1 WMMA.
__global__ __launch_bounds__(TPB, 1)
void decoder_persistent(const __bf16* __restrict__ wbf, const float* __restrict__ gi0,
                        const float* __restrict__ bhh0, const float* __restrict__ bih1,
                        const float* __restrict__ bhh1,
                        float* h0f, float* h1f, __bf16* h0b, __bf16* h1b,
                        unsigned* bar, float* __restrict__ out) {
  extern __shared__ char smem[];
  __bf16* slab = (__bf16*)smem;                          // 144 * PITCH bf16
  float*  accL = (float*)(smem + 144 * PITCH * 2);       // 36 jobs * 16 partials

  const int wg = blockIdx.x, tid = threadIdx.x;
  const int lane = tid & 31, wave = tid >> 5;
  const int c0 = wg * 16;

  // --- one-time weight preload into LDS ---
  // slab row sr = tile*16 + n ; tile 0-2: w_hh0 gate g ; 3-5: w_ih1 ; 6-8: w_hh1
  for (int sr = wave; sr < 144; sr += 8) {
    int tile = sr >> 4, nn = sr & 15;
    int mat = tile / 3, g = tile % 3;
    const __bf16* src = wbf + ((size_t)mat * 3 * C + g * C + c0 + nn) * C;
    __bf16* dst = slab + (size_t)sr * PITCH;
    for (int k = lane * 8; k < C; k += 32 * 8)
      *(v8bf*)(dst + k) = *(const v8bf*)(src + k);
  }
  __syncthreads();

  const int nsel   = lane & 15;
  const int koff16 = (lane & 16) ? 8 : 0;    // B: lanes 16-31 hold K 8..15 / 24..31
  const int abase  = (lane & 16) ? 8 : 0;    // A: lanes 16-31 hold K 8..15 / 24..31
  const int kq     = wave & 3;               // this wave's fixed k-quarter
  const int tpar   = wave >> 2;              // this wave's tile parity

  unsigned gen = 0;
  for (int t = -1; t < L; ++t) {
    if (t >= 0) {
      BfPack Ar[8];
      // --- section 1: tiles 0-5 (w_hh0, w_ih1), A = h0^{t+1} from global ---
      {
        const __bf16* hp = h0b + ((t + 1) & 1) * C + kq * 256 + abase;
        #pragma unroll
        for (int i = 0; i < 8; ++i) {
          Ar[i].h[0] = *(const v8bf*)(hp + i * 32);
          Ar[i].h[1] = *(const v8bf*)(hp + i * 32 + 16);
        }
        for (int tile = tpar; tile < 6; tile += 2) {
          const __bf16* rowp = slab + (size_t)(tile * 16 + nsel) * PITCH + kq * 256 + koff16;
          v8f dacc = {};
          #pragma unroll
          for (int i = 0; i < 8; ++i) {
            BfPack B;
            B.h[0] = *(const v8bf*)(rowp + i * 32);
            B.h[1] = *(const v8bf*)(rowp + i * 32 + 16);
            dacc = __builtin_amdgcn_wmma_f32_16x16x32_bf16(false, Ar[i].v, false, B.v,
                                                           (short)0, dacc, false, false);
          }
          if (lane < 16) accL[(tile * 4 + kq) * 16 + lane] = dacc[0];
        }
      }
      // --- section 2: tiles 6-8 (w_hh1), A = h1^{t} from global ---
      {
        const __bf16* hp = h1b + (t & 1) * C + kq * 256 + abase;
        #pragma unroll
        for (int i = 0; i < 8; ++i) {
          Ar[i].h[0] = *(const v8bf*)(hp + i * 32);
          Ar[i].h[1] = *(const v8bf*)(hp + i * 32 + 16);
        }
        for (int tile = 6 + tpar; tile < 9; tile += 2) {
          const __bf16* rowp = slab + (size_t)(tile * 16 + nsel) * PITCH + kq * 256 + koff16;
          v8f dacc = {};
          #pragma unroll
          for (int i = 0; i < 8; ++i) {
            BfPack B;
            B.h[0] = *(const v8bf*)(rowp + i * 32);
            B.h[1] = *(const v8bf*)(rowp + i * 32 + 16);
            dacc = __builtin_amdgcn_wmma_f32_16x16x32_bf16(false, Ar[i].v, false, B.v,
                                                           (short)0, dacc, false, false);
          }
          if (lane < 16) accL[(tile * 4 + kq) * 16 + lane] = dacc[0];
        }
      }
      __syncthreads();
    }

    // --- elementwise GRU update for this WG's 16 elements ---
    if (wave == 0 && lane < 16) {
      int c = c0 + lane;
      if (t < 0) {  // h0^1 from h0^0 = 0  (gh0 = b_hh0)
        float r = sigf(gi0[c] + bhh0[c]);
        float z = sigf(gi0[C + c] + bhh0[C + c]);
        float n = tanhf(gi0[2 * C + c] + r * bhh0[2 * C + c]);
        float h0n = (1.f - z) * n;
        h0f[1 * C + c] = h0n; h0b[1 * C + c] = (__bf16)h0n;
      } else {
        float sums[9];
        #pragma unroll
        for (int tl = 0; tl < 9; ++tl) {
          float v = 0.f;
          #pragma unroll
          for (int kk = 0; kk < 4; ++kk) v += accL[(tl * 4 + kk) * 16 + lane];
          sums[tl] = v;
        }
        float h0cur  = h0f[((t + 1) & 1) * C + c];
        float h1prev = h1f[(t & 1) * C + c];
        // layer0 -> h0^{t+2}
        float r0 = sigf(gi0[c] + sums[0] + bhh0[c]);
        float z0 = sigf(gi0[C + c] + sums[1] + bhh0[C + c]);
        float n0 = tanhf(gi0[2 * C + c] + r0 * (sums[2] + bhh0[2 * C + c]));
        float h0n = (1.f - z0) * n0 + z0 * h0cur;
        h0f[(t & 1) * C + c] = h0n; h0b[(t & 1) * C + c] = (__bf16)h0n;
        // layer1 -> h1^{t+1}
        float r1 = sigf(sums[3] + bih1[c] + sums[6] + bhh1[c]);
        float z1 = sigf(sums[4] + bih1[C + c] + sums[7] + bhh1[C + c]);
        float n1 = tanhf(sums[5] + bih1[2 * C + c] + r1 * (sums[8] + bhh1[2 * C + c]));
        float h1n = (1.f - z1) * n1 + z1 * h1prev;
        h1f[((t + 1) & 1) * C + c] = h1n; h1b[((t + 1) & 1) * C + c] = (__bf16)h1n;
        out[(size_t)t * C + c] = h1n;
        if (t == L - 1) {                         // previous_state tail: [h0^L, h1^L]
          out[(size_t)L * C + c]     = h0f[(L & 1) * C + c];
          out[(size_t)L * C + C + c] = h1n;
        }
      }
    }

    // --- device-wide barrier (monotonic counter, zeroed by memset each launch) ---
    ++gen;
    __syncthreads();
    if (tid == 0) {
      __threadfence();
      __hip_atomic_fetch_add(bar, 1u, __ATOMIC_RELEASE, __HIP_MEMORY_SCOPE_AGENT);
      while (__hip_atomic_load(bar, __ATOMIC_ACQUIRE, __HIP_MEMORY_SCOPE_AGENT) <
             (unsigned)NWG * gen)
        __builtin_amdgcn_s_sleep(1);
      __threadfence();
    }
    __syncthreads();
  }
}

// ---------------- host launch ----------------
extern "C" void kernel_launch(void* const* d_in, const int* in_sizes, int n_in,
                              void* d_out, int out_size, void* d_ws, size_t ws_size,
                              hipStream_t stream) {
  const float* x      = (const float*)d_in[0];
  const float* attn_w = (const float*)d_in[1];
  const float* attn_b = (const float*)d_in[2];
  const float* w_ih0  = (const float*)d_in[3];
  const float* w_hh0  = (const float*)d_in[4];
  const float* b_ih0  = (const float*)d_in[5];
  const float* b_hh0  = (const float*)d_in[6];
  const float* w_ih1  = (const float*)d_in[7];
  const float* w_hh1  = (const float*)d_in[8];
  const float* b_ih1  = (const float*)d_in[9];
  const float* b_hh1  = (const float*)d_in[10];
  float* out = (float*)d_out;
  char* ws = (char*)d_ws;

  const size_t WMAT = (size_t)3 * C * C;              // elements per weight matrix
  __bf16* wbf = (__bf16*)ws;                          // 3 matrices bf16 = 18.87 MB
  size_t SMALL = 3 * WMAT * sizeof(__bf16);
  float* s    = (float*)(ws + SMALL);                 // [L]
  float* a    = (float*)(ws + SMALL + 8192);          // [L]
  float* avf  = (float*)(ws + SMALL + 16384);         // [C]
  float* gi0  = (float*)(ws + SMALL + 24576);         // [3C]
  float* h0f  = (float*)(ws + SMALL + 36864);         // [2][C]
  float* h1f  = (float*)(ws + SMALL + 45056);         // [2][C]
  __bf16* h0b = (__bf16*)(ws + SMALL + 53248);        // [2][C]
  __bf16* h1b = (__bf16*)(ws + SMALL + 57344);        // [2][C]
  unsigned* bar = (unsigned*)(ws + SMALL + 61440);

  // zero barrier + h state (h^0 = 0) every launch -> deterministic replays
  hipMemsetAsync(ws + SMALL, 0, 65536, stream);

  // bf16 weight conversion (w_hh0, w_ih1, w_hh1)
  int n = (int)WMAT, cblocks = (n / 4 + 255) / 256;
  k_cvt<<<cblocks, 256, 0, stream>>>(w_hh0, wbf + 0 * WMAT, n);
  k_cvt<<<cblocks, 256, 0, stream>>>(w_ih1, wbf + 1 * WMAT, n);
  k_cvt<<<cblocks, 256, 0, stream>>>(w_hh1, wbf + 2 * WMAT, n);

  // attention (constant across timesteps) + gi0
  k_scores<<<L / 8, 256, 0, stream>>>(x, attn_w, attn_b, s);
  k_softmax<<<1, 256, 0, stream>>>(s, a);
  k_av<<<C / 256, 256, 0, stream>>>(x, a, avf);
  k_gi0<<<(3 * C) / 8, 256, 0, stream>>>(w_ih0, b_ih0, avf, gi0);
  (void)n_in; (void)in_sizes; (void)out_size; (void)ws_size;

  // persistent recurrence: 299,520 B dynamic LDS per WG (weights resident)
  size_t dynLds = (size_t)144 * PITCH * 2 + NJOBS * 16 * 4;
  hipFuncSetAttribute((const void*)decoder_persistent,
                      hipFuncAttributeMaxDynamicSharedMemorySize, (int)dynLds);
  decoder_persistent<<<NWG, TPB, dynLds, stream>>>(wbf, gi0, b_hh0, b_ih1, b_hh1,
                                                   h0f, h1f, h0b, h1b, bar, out);
}
